// MinJoint_33371895890309
// MI455X (gfx1250) — compile-verified, matrix operand order checked by reading, not run on hardware
//
#include <hip/hip_runtime.h>

// ---------------------------------------------------------------------------
// RNN-T joint network for MI455X (gfx1250, wave32, WMMA f32_16x16x32_f16).
//   enc  = enc_out  @ W_enc  + b_enc      (1024 x 512)
//   pred = pred_out @ W_pred + b_pred     ( 256 x 512)
//   out  = relu(enc[b,t,:] + pred[b,u,:]) @ W_out + b_out   (65536 x 1024)
// Main GEMM: 68.7 GFLOP; 268 MB f32 output -> ~11.5us HBM store floor at
// 23.3 TB/s, comparable to dense-f16 WMMA compute time. So: fuse add+ReLU
// into A-fragment build (64 MB joint tensor never hits memory), double-buffer
// W_out K-slabs in LDS via async global->LDS copies (ASYNCcnt), feed WMMA
// from ds_load_b128, and stream the output with non-temporal stores.
// ---------------------------------------------------------------------------

typedef __attribute__((ext_vector_type(16))) _Float16 v16h;
typedef __attribute__((ext_vector_type(8)))  _Float16 v8h;
typedef __attribute__((ext_vector_type(8)))  float    v8f;
typedef __attribute__((ext_vector_type(4)))  float    v4f;
typedef __attribute__((ext_vector_type(4)))  int      v4i;

#define MJ_AS1 __attribute__((address_space(1)))
#define MJ_AS3 __attribute__((address_space(3)))

#if __has_builtin(__builtin_amdgcn_global_load_async_to_lds_b128)
#define MJ_ASYNC 1
#else
#define MJ_ASYNC 0
#endif

// 16-byte global -> LDS copy (async when available; sync load+ds_store else).
__device__ __forceinline__ void mj_copy16_to_lds(const _Float16* src,
                                                 _Float16* dst) {
#if MJ_ASYNC
  __builtin_amdgcn_global_load_async_to_lds_b128((MJ_AS1 v4i*)src,
                                                 (MJ_AS3 v4i*)dst, 0, 0);
#else
  *(v8h*)dst = *(const v8h*)src;
#endif
}

__device__ __forceinline__ void mj_wait_async_le2() {
#if MJ_ASYNC
#if __has_builtin(__builtin_amdgcn_s_wait_asynccnt)
  __builtin_amdgcn_s_wait_asynccnt(2);
#else
  asm volatile("s_wait_asynccnt 2" ::: "memory");
#endif
#endif
}

__device__ __forceinline__ void mj_wait_async_le0() {
#if MJ_ASYNC
#if __has_builtin(__builtin_amdgcn_s_wait_asynccnt)
  __builtin_amdgcn_s_wait_asynccnt(0);
#else
  asm volatile("s_wait_asynccnt 0" ::: "memory");
#endif
#endif
}

// A-fragment (16x32 f16): joint row = relu(enc_row + pred_row), cvt to f16.
// Lane layout (ISA 7.12.2): lg = lane>>4, halves 0..7 -> K = kb+{0..7},
// halves 8..15 -> K = kb+16+{0..7}, kb = k0 + lg*8.
__device__ __forceinline__ v16h make_a_joint(const float* __restrict__ encRow,
                                             const float* __restrict__ predRow,
                                             int kb) {
  v4f e0 = *(const v4f*)(encRow + kb);
  v4f e1 = *(const v4f*)(encRow + kb + 4);
  v4f e2 = *(const v4f*)(encRow + kb + 16);
  v4f e3 = *(const v4f*)(encRow + kb + 20);
  v4f p0 = *(const v4f*)(predRow + kb);
  v4f p1 = *(const v4f*)(predRow + kb + 4);
  v4f p2 = *(const v4f*)(predRow + kb + 16);
  v4f p3 = *(const v4f*)(predRow + kb + 20);
  v16h a;
#pragma unroll
  for (int i = 0; i < 4; ++i) {
    a[i]      = (_Float16)fmaxf(e0[i] + p0[i], 0.0f);
    a[4 + i]  = (_Float16)fmaxf(e1[i] + p1[i], 0.0f);
    a[8 + i]  = (_Float16)fmaxf(e2[i] + p2[i], 0.0f);
    a[12 + i] = (_Float16)fmaxf(e3[i] + p3[i], 0.0f);
  }
  return a;
}

__device__ __forceinline__ v16h make_a_f32(const float* __restrict__ xRow,
                                           int kb) {
  v4f e0 = *(const v4f*)(xRow + kb);
  v4f e1 = *(const v4f*)(xRow + kb + 4);
  v4f e2 = *(const v4f*)(xRow + kb + 16);
  v4f e3 = *(const v4f*)(xRow + kb + 20);
  v16h a;
#pragma unroll
  for (int i = 0; i < 4; ++i) {
    a[i]      = (_Float16)e0[i];
    a[4 + i]  = (_Float16)e1[i];
    a[8 + i]  = (_Float16)e2[i];
    a[12 + i] = (_Float16)e3[i];
  }
  return a;
}

// B-fragment (32x16) straight from global [N][K] f16 (projection kernels).
__device__ __forceinline__ v16h load_b_frag(const _Float16* __restrict__ Wt,
                                            int n, int K, int kb) {
  const _Float16* p = Wt + (size_t)n * K + kb;
  v8h lo = *(const v8h*)(p);
  v8h hi = *(const v8h*)(p + 16);
  v16h r;
#pragma unroll
  for (int i = 0; i < 8; ++i) { r[i] = lo[i]; r[i + 8] = hi[i]; }
  return r;
}

// ---------------------------------------------------------------------------
// Weight transpose + f32->f16 convert:  W[R][C] -> Wt[C][R]
// ---------------------------------------------------------------------------
__global__ void mj_transpose_h(const float* __restrict__ W,
                               _Float16* __restrict__ Wt, int R, int C) {
  int i = blockIdx.x * blockDim.x + threadIdx.x;
  if (i >= R * C) return;
  int r = i / C, c = i % C;  // coalesced read (c fastest)
  Wt[(size_t)c * R + r] = (_Float16)W[i];
}

// ---------------------------------------------------------------------------
// Projection GEMM: Y[M][512] = X[M][K] * W + bias, f32 out.
// One wave per 16x64 tile; Wt is [512][K] f16.
// ---------------------------------------------------------------------------
__global__ __launch_bounds__(32)
void mj_proj_gemm(const float* __restrict__ X, const _Float16* __restrict__ Wt,
                  const float* __restrict__ bias, float* __restrict__ Y,
                  int K) {
  const int lane = threadIdx.x & 31;
  const int lg = lane >> 4, ln = lane & 15;
  const int m0 = blockIdx.x * 16;
  const int nbase = blockIdx.y * 64;

  const float* xRow = X + (size_t)(m0 + ln) * K;
  v8f c[4] = {};

  for (int k0 = 0; k0 < K; k0 += 32) {
    const int kb = k0 + lg * 8;
    v16h a = make_a_f32(xRow, kb);
#pragma unroll
    for (int j = 0; j < 4; ++j) {
      v16h bf = load_b_frag(Wt, nbase + j * 16 + ln, K, kb);
      c[j] = __builtin_amdgcn_wmma_f32_16x16x32_f16(
          false, a, false, bf, (short)0, c[j], false, false);
    }
  }

#pragma unroll
  for (int j = 0; j < 4; ++j) {
    const int n = nbase + j * 16 + ln;
    const float bv = bias[n];
#pragma unroll
    for (int r = 0; r < 8; ++r) {
      const int m = m0 + r + 8 * lg;  // C/D layout: row = vgpr + 8*lanegrp
      Y[(size_t)m * 512 + n] = c[j][r] + bv;
    }
  }
}

// ---------------------------------------------------------------------------
// Fused joint GEMM.  Block = 256 thr (8 waves): one (b,t), 128-wide V slab.
// Waves: 4 U-tiles x 2 V-halves; each wave -> 16(U) x 64(V), 4 accumulators.
// W_out K-slabs (32 K x 128 N f16) double-buffered in LDS via async copies.
// LDS rows padded to 40 halves (80B): lane bank stride 20 -> conflict-free
// ds_load_b128 across the 16 lanes of each half-wave.
// ---------------------------------------------------------------------------
#define MJ_BROW 40  // padded halves per N row in LDS slab

__global__ __launch_bounds__(256)
void mj_joint_gemm(const float* __restrict__ enc,       // [1024][512]
                   const float* __restrict__ pred,      // [256][512]
                   const _Float16* __restrict__ WoutT,  // [1024][512] f16
                   const float* __restrict__ b_out,     // [1024]
                   float* __restrict__ out) {           // [1024][64][1024]
  __shared__ __align__(16) _Float16 ldsB[2][128 * MJ_BROW];

  const int tid = threadIdx.x;
  const int lane = tid & 31, wid = tid >> 5;
  const int lg = lane >> 4, ln = lane & 15;
  const int u0 = (wid >> 1) * 16;        // 0,16,32,48
  const int nloc0 = (wid & 1) * 64;      // wave's V half within slab
  const int nbase = blockIdx.x * 128;
  const int bt = blockIdx.y;             // b*256 + t
  const int b = bt >> 8;

  const float* encRow = enc + (size_t)bt * 512;
  const float* predRow = pred + (size_t)(b * 64 + u0 + ln) * 512;
  const _Float16* wsrc = WoutT + (size_t)nbase * 512;

  // This thread's two 16B staging chunks: e -> (n = e>>2, q = e&3)
  const int sn0 = tid >> 2, sq0 = (tid & 3) * 8;
  const int sn1 = (tid + 256) >> 2, sq1 = ((tid + 256) & 3) * 8;

  // Prologue: stage slab k=0 into buffer 0.
  mj_copy16_to_lds(wsrc + (size_t)sn0 * 512 + sq0, &ldsB[0][sn0 * MJ_BROW + sq0]);
  mj_copy16_to_lds(wsrc + (size_t)sn1 * 512 + sq1, &ldsB[0][sn1 * MJ_BROW + sq1]);

  v8f c[4] = {};
  int buf = 0;
  for (int i = 0; i < 16; ++i) {
    const int k0 = i * 32;
    if (i < 15) {  // prefetch next slab into the other buffer
      const int kn = k0 + 32;
      mj_copy16_to_lds(wsrc + (size_t)sn0 * 512 + kn + sq0,
                       &ldsB[buf ^ 1][sn0 * MJ_BROW + sq0]);
      mj_copy16_to_lds(wsrc + (size_t)sn1 * 512 + kn + sq1,
                       &ldsB[buf ^ 1][sn1 * MJ_BROW + sq1]);
      mj_wait_async_le2();  // oldest 2 (current buf) complete, prefetch flies
    } else {
      mj_wait_async_le0();
    }
    __syncthreads();

    const int kb = k0 + lg * 8;
    v16h a = make_a_joint(encRow, predRow, kb);
#pragma unroll
    for (int j = 0; j < 4; ++j) {
      const _Float16* bp =
          &ldsB[buf][(nloc0 + j * 16 + ln) * MJ_BROW + lg * 8];
      v8h lo = *(const v8h*)bp;          // ds_load_b128
      v8h hi = *(const v8h*)(bp + 16);   // ds_load_b128
      v16h bf;
#pragma unroll
      for (int q = 0; q < 8; ++q) { bf[q] = lo[q]; bf[q + 8] = hi[q]; }
      c[j] = __builtin_amdgcn_wmma_f32_16x16x32_f16(
          false, a, false, bf, (short)0, c[j], false, false);
    }
    __syncthreads();
    buf ^= 1;
  }

  // Epilogue: bias + non-temporal streaming stores (268 MB, keep L2 clean).
#pragma unroll
  for (int j = 0; j < 4; ++j) {
    const int n = nbase + nloc0 + j * 16 + ln;
    const float bv = b_out[n];
#pragma unroll
    for (int r = 0; r < 8; ++r) {
      const int u = u0 + r + 8 * lg;
      __builtin_nontemporal_store(c[j][r] + bv,
                                  &out[((size_t)bt * 64 + u) * 1024 + n]);
    }
  }
}

// ---------------------------------------------------------------------------
extern "C" void kernel_launch(void* const* d_in, const int* in_sizes, int n_in,
                              void* d_out, int out_size, void* d_ws,
                              size_t ws_size, hipStream_t stream) {
  const float* enc_in  = (const float*)d_in[0];  // (4,256,512)
  const float* pred_in = (const float*)d_in[1];  // (4,64,640)
  const float* W_enc   = (const float*)d_in[2];  // (512,512)
  const float* b_enc   = (const float*)d_in[3];  // (512)
  const float* W_pred  = (const float*)d_in[4];  // (640,512)
  const float* b_pred  = (const float*)d_in[5];  // (512)
  const float* W_out   = (const float*)d_in[6];  // (512,1024)
  const float* b_out   = (const float*)d_in[7];  // (1024)
  float* out = (float*)d_out;                    // (4,256,64,1024)

  // Workspace: WoutT f16 1MB | WencT f16 512KB | WpredT f16 640KB |
  //            encP f32 2MB | predP f32 512KB   (~4.7MB total)
  char* ws = (char*)d_ws;
  _Float16* WoutT  = (_Float16*)(ws);
  _Float16* WencT  = (_Float16*)(ws + (1u << 20));
  _Float16* WpredT = (_Float16*)(ws + (1u << 20) + 512 * 512 * 2);
  float* encP  = (float*)(ws + (1u << 20) + 512 * 512 * 2 + 512 * 640 * 2);
  float* predP = encP + 1024 * 512;

  mj_transpose_h<<<(512 * 1024 + 255) / 256, 256, 0, stream>>>(W_out, WoutT,
                                                               512, 1024);
  mj_transpose_h<<<(512 * 512 + 255) / 256, 256, 0, stream>>>(W_enc, WencT,
                                                              512, 512);
  mj_transpose_h<<<(640 * 512 + 255) / 256, 256, 0, stream>>>(W_pred, WpredT,
                                                              640, 512);

  mj_proj_gemm<<<dim3(1024 / 16, 512 / 64), 32, 0, stream>>>(enc_in, WencT,
                                                             b_enc, encP, 512);
  mj_proj_gemm<<<dim3(256 / 16, 512 / 64), 32, 0, stream>>>(pred_in, WpredT,
                                                            b_pred, predP, 640);

  mj_joint_gemm<<<dim3(1024 / 128, 4 * 256), 256, 0, stream>>>(
      encP, predP, WoutT, b_out, out);
}
